// CrossAttention_50070728736956
// MI455X (gfx1250) — matrix operand order repaired and checked by reading, not compiled, not run on hardware
//
#include <hip/hip_runtime.h>

// ---------------------------------------------------------------------------
// CDNA5 (gfx1250) cross-attention (MQA), bf16 WMMA pipeline:
//   1) cvt x,k,v -> bf16 ; transpose-cvt Wq,Wo -> bf16 [N][K]
//   2) q = x @ Wq      (WMMA bf16, double-buffered async LDS tiles)
//   3) flash attention (WMMA bf16 QK^T and PV, online softmax, j-tile 64)
//   4) out = ao @ Wo   (WMMA bf16, f32 output)
// ---------------------------------------------------------------------------

typedef __bf16 bf16;
typedef __attribute__((ext_vector_type(16))) __bf16 v16bf;
typedef __attribute__((ext_vector_type(8)))  __bf16 v8bf;
typedef __attribute__((ext_vector_type(4)))  __bf16 v4bf;
typedef __attribute__((ext_vector_type(8)))  float  v8f;
typedef __attribute__((ext_vector_type(4)))  int    v4i;

#define DIM_   2048
#define HEADS_ 16
#define DHEAD_ 128
#define B_     2
#define N_     2048
#define J_     2048
#define INNER_ (HEADS_ * DHEAD_)   // 2048
#define MROWS_ (B_ * N_)           // 4096

// ---- gfx1250 async global->LDS (guarded; falls back to sync copy) ----------

#if defined(__has_builtin)
#if __has_builtin(__builtin_amdgcn_global_load_async_to_lds_b128)
#define HAVE_ASYNC_LDS 1
#endif
#if __has_builtin(__builtin_amdgcn_s_wait_asynccnt)
#define HAVE_WAIT_ASYNC 1
#endif
#endif

__device__ __forceinline__ void async_copy_b128(const bf16* g, bf16* l) {
#ifdef HAVE_ASYNC_LDS
  __builtin_amdgcn_global_load_async_to_lds_b128((v4i*)(void*)g,
                                                 (v4i*)(void*)l, 0, 0);
#else
  *(uint4*)l = *(const uint4*)g;
#endif
}

__device__ __forceinline__ void wait_async() {
#ifdef HAVE_ASYNC_LDS
#ifdef HAVE_WAIT_ASYNC
  __builtin_amdgcn_s_wait_asynccnt(0);
#else
  asm volatile("s_wait_asynccnt 0x0" ::: "memory");
#endif
#endif
}

// ---- helpers ---------------------------------------------------------------

__device__ __forceinline__ v16bf ld16_split(const bf16* lo, const bf16* hi) {
  v8bf a = *(const v8bf*)lo;
  v8bf b = *(const v8bf*)hi;
  v16bf r;
#pragma unroll
  for (int i = 0; i < 8; ++i) { r[i] = a[i]; r[i + 8] = b[i]; }
  return r;
}
__device__ __forceinline__ v16bf ld16(const bf16* p) { return ld16_split(p, p + 8); }

__device__ __forceinline__ v8f wmma_bf16(v16bf a, v16bf b, v8f c) {
  return __builtin_amdgcn_wmma_f32_16x16x32_bf16(false, a, false, b, (short)0, c,
                                                 false, false);
}

// ---- 1) elementwise f32 -> bf16 -------------------------------------------

__global__ void cvt_f32_bf16(const float* __restrict__ s, bf16* __restrict__ d, int n4) {
  int i = blockIdx.x * blockDim.x + threadIdx.x;
  if (i < n4) {
    float4 f = ((const float4*)s)[i];
    v4bf o;
    o[0] = (bf16)f.x; o[1] = (bf16)f.y; o[2] = (bf16)f.z; o[3] = (bf16)f.w;
    ((v4bf*)d)[i] = o;
  }
}

// ---- 1b) W [K][N] f32  ->  Wt [N][K] bf16 ---------------------------------

__global__ void transpose_cvt(const float* __restrict__ W, bf16* __restrict__ Wt,
                              int K, int Ncols) {
  __shared__ float tile[32][33];
  int tx = threadIdx.x;        // 0..31
  int ty = threadIdx.y;        // 0..7
  int kbase = blockIdx.y * 32;
  int nbase = blockIdx.x * 32;
#pragma unroll
  for (int r = 0; r < 32; r += 8)
    tile[ty + r][tx] = W[(size_t)(kbase + ty + r) * Ncols + nbase + tx];
  __syncthreads();
#pragma unroll
  for (int r = 0; r < 32; r += 8)
    Wt[(size_t)(nbase + ty + r) * K + kbase + tx] = (bf16)tile[tx][ty + r];
}

// ---- 2/4) GEMM: C[M,N] = A[M,K](bf16 rowmajor) x Bt[N,K](bf16 rowmajor)^T --
// Double-buffered LDS K-slices filled with async global->LDS copies.
// MODE 0: C = f32 [M][N]   (final projection)
// MODE 1: C = bf16, scaled, relayout to q[b][h][n][d]

template <int MODE>
__global__ void __launch_bounds__(256)
gemm_bf16(const bf16* __restrict__ A, const bf16* __restrict__ Bt,
          void* __restrict__ Cout, int Mdim, int Ndim, int Kdim, float scale) {
  __shared__ bf16 As[2][128 * 32];   // [m][k]
  __shared__ bf16 Bs[2][128 * 32];   // [n][k]

  const int tid  = threadIdx.x;
  const int lane = tid & 31;
  const int wave = tid >> 5;      // 0..7
  const int wy   = wave & 3;      // 4 waves along M (32 rows each)
  const int wx   = wave >> 2;     // 2 waves along N (64 cols each)
  const int l16  = lane & 15;
  const int hs   = lane >> 4;     // half-wave select
  const int m0   = blockIdx.y * 128;
  const int n0   = blockIdx.x * 128;

  const int crow = tid >> 1;            // 0..127: cooperative-copy row
  const int cel  = (tid & 1) * 16;      // element offset within 32-wide row

  v8f acc[2][4] = {};

  // issue one K-slice (each thread: 2x16B for A, 2x16B for B, async)
  auto issue = [&](int kc, int b) {
    const bf16* gA = A + (size_t)(m0 + crow) * Kdim + kc + cel;
    bf16* sA = &As[b][crow * 32 + cel];
    async_copy_b128(gA, sA);
    async_copy_b128(gA + 8, sA + 8);
    const bf16* gB = Bt + (size_t)(n0 + crow) * Kdim + kc + cel;
    bf16* sB = &Bs[b][crow * 32 + cel];
    async_copy_b128(gB, sB);
    async_copy_b128(gB + 8, sB + 8);
  };

  int buf = 0;
  issue(0, 0);
  wait_async();
  __syncthreads();

  for (int kc = 0; kc < Kdim; kc += 32) {
    if (kc + 32 < Kdim) issue(kc + 32, buf ^ 1);   // overlap with compute

    v16bf afr[2];
#pragma unroll
    for (int mi = 0; mi < 2; ++mi) {
      const bf16* p = &As[buf][(32 * wy + 16 * mi + l16) * 32 + hs * 8];
      afr[mi] = ld16_split(p, p + 16);       // K {0..7,16..23} / {8..15,24..31}
    }
    v16bf bfr[4];
#pragma unroll
    for (int ni = 0; ni < 4; ++ni) {
      const bf16* p = &Bs[buf][(64 * wx + 16 * ni + l16) * 32 + hs * 8];
      bfr[ni] = ld16_split(p, p + 16);       // K {0..7,16..23} / {8..15,24..31}
    }
#pragma unroll
    for (int mi = 0; mi < 2; ++mi)
#pragma unroll
      for (int ni = 0; ni < 4; ++ni)
        acc[mi][ni] = wmma_bf16(afr[mi], bfr[ni], acc[mi][ni]);

    wait_async();        // next slice resident
    __syncthreads();     // everyone done reading buf / writing buf^1
    buf ^= 1;
  }

  // epilogue (C layout: row = v + 8*hs, col = l16)
#pragma unroll
  for (int mi = 0; mi < 2; ++mi)
#pragma unroll
    for (int ni = 0; ni < 4; ++ni)
#pragma unroll
      for (int r = 0; r < 8; ++r) {
        int grow = m0 + 32 * wy + 16 * mi + r + 8 * hs;
        int gcol = n0 + 64 * wx + 16 * ni + l16;
        float val = acc[mi][ni][r] * scale;
        if (MODE == 0) {
          ((float*)Cout)[(size_t)grow * Ndim + gcol] = val;
        } else {
          int b = grow >> 11, n = grow & 2047;
          int h = gcol >> 7,  d = gcol & 127;
          ((bf16*)Cout)[(((size_t)(b * HEADS_ + h) * N_) + n) * DHEAD_ + d] = (bf16)val;
        }
      }
}

// ---- 3) flash attention (MQA), j-tile 64 ----------------------------------
// grid: (N/128, HEADS, B), block: 256 (8 waves x 16 q-rows)

__global__ void __launch_bounds__(256)
flash_attn(const bf16* __restrict__ q, const bf16* __restrict__ k,
           const bf16* __restrict__ v, bf16* __restrict__ ao) {
  __shared__ bf16 k_lds[64 * 128];     // [j][d]
  __shared__ bf16 v_stage[64 * 128];   // [j][d] (async landing zone)
  __shared__ bf16 vT_lds[128 * 64];    // [d][j]
  __shared__ bf16 p_lds[8 * 16 * 64];  // per-wave C->A relayout staging

  const int tid  = threadIdx.x;
  const int lane = tid & 31;
  const int wave = tid >> 5;
  const int l16  = lane & 15;
  const int hs   = lane >> 4;
  const int b    = blockIdx.z;
  const int h    = blockIdx.y;
  const int qbase = blockIdx.x * 128 + wave * 16;

  // q fragments for this wave's 16 rows, K=128 -> 4 chunks (A layout)
  const bf16* qrow =
      q + (((size_t)(b * HEADS_ + h) * N_) + qbase + l16) * DHEAD_;
  v16bf qf[4];
#pragma unroll
  for (int c = 0; c < 4; ++c) {
    const bf16* p = qrow + c * 32 + hs * 8;
    qf[c] = ld16_split(p, p + 16);
  }

  v8f of[8] = {};                      // out accumulators, N = 8 x 16 (d = 128)
  float mrow[8], lrow[8];
#pragma unroll
  for (int r = 0; r < 8; ++r) { mrow[r] = -3.0e38f; lrow[r] = 0.0f; }

  const bf16* kb = k + (size_t)b * J_ * DHEAD_;
  const bf16* vb = v + (size_t)b * J_ * DHEAD_;
  bf16* pw = p_lds + wave * (16 * 64);

  const int cj  = tid >> 2;            // 0..63 cooperative row
  const int cdb = (tid & 3) * 32;      // 32-element segment base

  for (int j0 = 0; j0 < J_; j0 += 64) {
    __syncthreads();   // previous iteration's readers are done

    // async k tile + v staging tile (64B per thread each)
    {
      const bf16* gk = kb + (size_t)(j0 + cj) * DHEAD_ + cdb;
      const bf16* gv = vb + (size_t)(j0 + cj) * DHEAD_ + cdb;
      bf16* sk = k_lds + cj * DHEAD_ + cdb;
      bf16* sv = v_stage + cj * DHEAD_ + cdb;
#pragma unroll
      for (int u = 0; u < 4; ++u) {
        async_copy_b128(gk + u * 8, sk + u * 8);
        async_copy_b128(gv + u * 8, sv + u * 8);
      }
    }
    wait_async();
    __syncthreads();

    // LDS->LDS transpose: vT[d][j] = v_stage[j][d]
#pragma unroll
    for (int e = 0; e < 32; ++e)
      vT_lds[(cdb + e) * 64 + cj] = v_stage[cj * DHEAD_ + cdb + e];
    __syncthreads();

    // sim = q . k^T  (four 16-wide j fragments, K=128 over 4 chunks)
    v8f sf[4] = {};
#pragma unroll
    for (int jf = 0; jf < 4; ++jf)
#pragma unroll
      for (int c = 0; c < 4; ++c) {
        const bf16* p = k_lds + (jf * 16 + l16) * DHEAD_ + c * 32 + hs * 8;
        sf[jf] = wmma_bf16(qf[c], ld16_split(p, p + 16), sf[jf]);
      }

    // online softmax per row (row r lives on vgpr r across a 16-lane half)
#pragma unroll
    for (int r = 0; r < 8; ++r) {
      float mx = fmaxf(fmaxf(sf[0][r], sf[1][r]), fmaxf(sf[2][r], sf[3][r]));
#pragma unroll
      for (int m = 1; m < 16; m <<= 1) mx = fmaxf(mx, __shfl_xor(mx, m, 32));
      float nm   = fmaxf(mrow[r], mx);
      float corr = __expf(mrow[r] - nm);
      float rs = 0.0f;
#pragma unroll
      for (int jf = 0; jf < 4; ++jf) {
        float p = __expf(sf[jf][r] - nm);
        sf[jf][r] = p;
        rs += p;
      }
#pragma unroll
      for (int m = 1; m < 16; m <<= 1) rs += __shfl_xor(rs, m, 32);
      lrow[r] = lrow[r] * corr + rs;
      mrow[r] = nm;
#pragma unroll
      for (int nf = 0; nf < 8; ++nf) of[nf][r] *= corr;
    }

    // stage p through per-wave LDS: C layout -> A layout
#pragma unroll
    for (int r = 0; r < 8; ++r) {
      int row = r + 8 * hs;
#pragma unroll
      for (int jf = 0; jf < 4; ++jf)
        pw[row * 64 + jf * 16 + l16] = (bf16)sf[jf][r];
    }
    asm volatile("s_wait_dscnt 0" ::: "memory");
    v16bf pa[2];
#pragma unroll
    for (int jc = 0; jc < 2; ++jc) {
      const bf16* pr = pw + l16 * 64 + jc * 32 + hs * 8;
      pa[jc] = ld16_split(pr, pr + 16);
    }

    // out += p @ v  (8 d-fragments x 2 j-chunks)
#pragma unroll
    for (int nf = 0; nf < 8; ++nf)
#pragma unroll
      for (int jc = 0; jc < 2; ++jc) {
        const bf16* p = vT_lds + (nf * 16 + l16) * 64 + jc * 32 + hs * 8;
        of[nf] = wmma_bf16(pa[jc], ld16_split(p, p + 16), of[nf]);
      }
  }

  // normalize and store attention output as ao[b*N+n][h*128+d] (bf16)
#pragma unroll
  for (int r = 0; r < 8; ++r) {
    float inv = 1.0f / lrow[r];
    int n = qbase + r + 8 * hs;
#pragma unroll
    for (int nf = 0; nf < 8; ++nf) {
      int d = nf * 16 + l16;
      ao[((size_t)b * N_ + n) * INNER_ + h * DHEAD_ + d] = (bf16)(of[nf][r] * inv);
    }
  }
}

// ---- host side -------------------------------------------------------------

extern "C" void kernel_launch(void* const* d_in, const int* in_sizes, int n_in,
                              void* d_out, int out_size, void* d_ws, size_t ws_size,
                              hipStream_t stream) {
  (void)in_sizes; (void)n_in; (void)out_size; (void)ws_size;
  const float* x  = (const float*)d_in[0];
  const float* k  = (const float*)d_in[1];
  const float* v  = (const float*)d_in[2];
  const float* Wq = (const float*)d_in[3];
  const float* Wo = (const float*)d_in[4];
  float* out = (float*)d_out;

  bf16* x_bf = (bf16*)d_ws;                               // 4096*2048
  bf16* wqt  = x_bf + (size_t)MROWS_ * DIM_;              // 2048*2048 (W^T)
  bf16* wot  = wqt  + (size_t)DIM_ * INNER_;              // 2048*2048 (W^T)
  bf16* k_bf = wot  + (size_t)INNER_ * DIM_;              // 2*2048*128
  bf16* v_bf = k_bf + (size_t)B_ * J_ * DHEAD_;
  bf16* q_bf = v_bf + (size_t)B_ * J_ * DHEAD_;           // [b][h][n][d]
  bf16* ao   = q_bf + (size_t)B_ * HEADS_ * N_ * DHEAD_;  // [b*n][inner]

  const int xN4 = (MROWS_ * DIM_) / 4;
  const int kN4 = (B_ * J_ * DHEAD_) / 4;
  cvt_f32_bf16<<<(xN4 + 255) / 256, 256, 0, stream>>>(x, x_bf, xN4);
  cvt_f32_bf16<<<(kN4 + 255) / 256, 256, 0, stream>>>(k, k_bf, kN4);
  cvt_f32_bf16<<<(kN4 + 255) / 256, 256, 0, stream>>>(v, v_bf, kN4);

  transpose_cvt<<<dim3(INNER_ / 32, DIM_ / 32), dim3(32, 8), 0, stream>>>(
      Wq, wqt, DIM_, INNER_);
  transpose_cvt<<<dim3(DIM_ / 32, INNER_ / 32), dim3(32, 8), 0, stream>>>(
      Wo, wot, INNER_, DIM_);

  const float scale = 0.08838834764831845f;  // DIM_HEAD^-0.5
  gemm_bf16<1><<<dim3(INNER_ / 128, MROWS_ / 128), 256, 0, stream>>>(
      x_bf, wqt, (void*)q_bf, MROWS_, INNER_, DIM_, scale);

  flash_attn<<<dim3(N_ / 128, HEADS_, B_), 256, 0, stream>>>(q_bf, k_bf, v_bf, ao);

  gemm_bf16<0><<<dim3(DIM_ / 128, MROWS_ / 128), 256, 0, stream>>>(
      ao, wot, (void*)out, MROWS_, DIM_, INNER_, 1.0f);
}
